// Block_19207093748096
// MI455X (gfx1250) — matrix operand order repaired and checked by reading, not compiled, not run on hardware
//
#include <hip/hip_runtime.h>
#include <hip/hip_bf16.h>
#include <math.h>

// ---------------------------------------------------------------------------
// CDNA5 (gfx1250) implementation of the attention+MoE transformer block.
// All GEMM work goes through V_WMMA_F32_16X16X32_BF16 (wave32). GEMM K-loop is
// LDS double-buffered: A tiles prefetched with global_load_async_to_lds_b128
// (ASYNCcnt) into the alternate buffer while the current tile is consumed;
// B tiles staged in fragment-ready LDS layout so fragments are ds_load_b128.
// ---------------------------------------------------------------------------

typedef __bf16 bf16;
typedef __attribute__((ext_vector_type(16))) __bf16 v16bf;
typedef __attribute__((ext_vector_type(8)))  float    v8f;

union FragU { v16bf v; uint4 q[2]; };

static __device__ __forceinline__ v8f wmma_bf16(v16bf a, v16bf b, v8f c) {
  // (neg_a, A, neg_b, B, c_mod, C, reuse_a, reuse_b)
  return __builtin_amdgcn_wmma_f32_16x16x32_bf16(false, a, false, b, (short)0, c,
                                                 false, false);
}

// ===========================================================================
// Generic tiled GEMM:  C[M,N](f32 or bf16) = op(A[M,K] * B[K,N]) (+ residual)
// A,B bf16 row-major. Block tile 64x128, K-step 32, 8 waves -> each wave 32x32.
// flags: bit0 = exact GELU, bit1 = store bf16, bit2 = add f32 residual.
// Requires M%64==0, N%128==0, K%32==0 (true for every GEMM in this model).
//
// Dynamic LDS: two 13312 B buffers, each:
//   [0,     5120) : A tile, 64 rows x 40 bf16 (row stride 80 B, 16B aligned)
//   [5120, 13312) : B tile in fragment layout: [8 colGroups][32 lanes][16 bf16]
//                   chunk(group,lane) holds B[kh*16 + t][colGroup*16 + n] with
//                   lane = n + 16*kh, t = 0..15  -> one 32B run per lane.
// ===========================================================================
#define GEMM_LDS_HALF  13312
#define GEMM_LDS_BYTES (2 * GEMM_LDS_HALF)

__global__ __launch_bounds__(256) void gemm_bf16_wmma_kernel(
    const bf16* __restrict__ A, const bf16* __restrict__ B,
    void* __restrict__ C, const float* __restrict__ Res,
    int M, int N, int K,
    long batchStrideA, long batchStrideB, long batchStrideC, long batchStrideRes,
    int flags)
{
  extern __shared__ char smem[];

  const int lane  = threadIdx.x & 31;
  const int wv    = threadIdx.x >> 5;
  const int waveM = wv >> 2;           // 0..1
  const int waveN = wv & 3;            // 0..3
  const int rowBase = blockIdx.y * 64;
  const int colBase = blockIdx.x * 128;
  const long zb = blockIdx.z;
  const bf16* Ab = A + zb * batchStrideA;
  const bf16* Bb = B + zb * batchStrideB;

  const int mrow  = lane & 15;
  const int khalf = lane >> 4;

  // per-thread staging coordinates
  const int ar = threadIdx.x >> 2;           // A row 0..63
  const int ac = (threadIdx.x & 3) * 8;      // A col 0,8,16,24
  const int br = threadIdx.x >> 3;           // B row (kk) 0..31
  const int bc = (threadIdx.x & 7) * 16;     // B col 0..112
  const int bScatterOff =                    // within a buffer's B region
      (((bc >> 4) * 32 + ((br >> 4) << 4)) * 16) + (br & 15);

  auto stageA_async = [&](int k0, int buf) {
    const bf16* src = Ab + (long)(rowBase + ar) * K + k0 + ac;
    const unsigned dst = (unsigned)(buf * GEMM_LDS_HALF + ar * 80 + ac * 2);
    asm volatile("global_load_async_to_lds_b128 %0, %1, off"
                 :: "v"(dst), "v"((unsigned long long)(uintptr_t)src)
                 : "memory");
  };
  auto loadBrow = [&](int k0, uint4& q0, uint4& q1) {
    const uint4* src = (const uint4*)(Bb + (long)(k0 + br) * N + colBase + bc);
    q0 = src[0];
    q1 = src[1];
  };
  auto scatterB = [&](int buf, uint4 q0, uint4 q1) {
    bf16 vals[16];
    *(uint4*)&vals[0] = q0;
    *(uint4*)&vals[8] = q1;
    bf16* dstBase = (bf16*)(smem + buf * GEMM_LDS_HALF + 5120) + bScatterOff;
#pragma unroll
    for (int e = 0; e < 16; ++e)
      dstBase[e * 16] = vals[e];              // lane stride = 16 bf16 = 32 B
  };

  v8f acc[2][2] = {};

  // prologue: stage K-step 0 into buffer 0
  {
    uint4 q0, q1;
    stageA_async(0, 0);
    loadBrow(0, q0, q1);
    scatterB(0, q0, q1);
    asm volatile("s_wait_asynccnt 0" ::: "memory");
    __syncthreads();
  }

  int buf = 0;
  for (int k0 = 0; k0 < K; k0 += 32) {
    const int  kn      = k0 + 32;
    const bool hasNext = kn < K;

    // prefetch next tile: async A into the alternate buffer, B rows into regs
    uint4 nq0, nq1;
    if (hasNext) {
      stageA_async(kn, buf ^ 1);
      loadBrow(kn, nq0, nq1);
    }

    // --- consume current buffer: contiguous 128-bit LDS fragment loads
    const bf16* As  = (const bf16*)(smem + buf * GEMM_LDS_HALF);
    const bf16* BsF = (const bf16*)(smem + buf * GEMM_LDS_HALF + 5120);
    FragU aF[2], bF[2];
#pragma unroll
    for (int i = 0; i < 2; ++i) {
      const bf16* arow = As + (waveM * 32 + i * 16 + mrow) * 40 + khalf * 16;
      aF[i].q[0] = *(const uint4*)(arow);        // K = kh*16 .. +7
      aF[i].q[1] = *(const uint4*)(arow + 8);    // K = kh*16+8 .. +7
    }
#pragma unroll
    for (int j = 0; j < 2; ++j) {
      const uint4* bp =
          (const uint4*)(BsF + ((waveN * 2 + j) * 32 + lane) * 16);
      bF[j].q[0] = bp[0];
      bF[j].q[1] = bp[1];
    }
#pragma unroll
    for (int i = 0; i < 2; ++i)
#pragma unroll
      for (int j = 0; j < 2; ++j)
        acc[i][j] = wmma_bf16(aF[i].v, bF[j].v, acc[i][j]);

    // --- finish staging the next buffer, then swap
    if (hasNext) {
      scatterB(buf ^ 1, nq0, nq1);   // loadcnt wait lands here, after the WMMAs
      asm volatile("s_wait_asynccnt 0" ::: "memory");
      __syncthreads();
      buf ^= 1;
    }
  }

  const bool doGelu  = (flags & 1) != 0;
  const bool outBf16 = (flags & 2) != 0;
  const bool addRes  = (flags & 4) != 0;
#pragma unroll
  for (int i = 0; i < 2; ++i) {
#pragma unroll
    for (int j = 0; j < 2; ++j) {
#pragma unroll
      for (int r = 0; r < 8; ++r) {
        const int m = rowBase + waveM * 32 + i * 16 + r + 8 * khalf;
        const int n = colBase + waveN * 32 + j * 16 + mrow;
        float v = acc[i][j][r];
        if (doGelu) v = 0.5f * v * (1.0f + erff(v * 0.70710678118654752440f));
        if (addRes) v += Res[zb * batchStrideRes + (long)m * N + n];
        const long off = zb * batchStrideC + (long)m * N + n;
        if (outBf16) ((bf16*)C)[off] = (bf16)v;
        else         ((float*)C)[off] = v;
      }
    }
  }
}

// ===========================================================================
// Flash-attention: one wave per (b, head, 16-row q tile). hd = 64.
// Online softmax over 32-wide KV chunks; QK^T and P*V through WMMA.
// ===========================================================================
__global__ __launch_bounds__(256) void attn_kernel(
    const bf16* __restrict__ qkv,   // [B*T, 3C] (q | k | v)
    bf16* __restrict__ y,           // [B*T, C]
    int T, int Cdim, int nh)
{
  const int hd = 64;
  __shared__ bf16 Pb[8][16][40];    // per-wave P staging (16x32, padded rows)

  const int lane = threadIdx.x & 31;
  const int wv   = threadIdx.x >> 5;
  const int tile = blockIdx.x * 8 + wv;
  const int tilesPerHead = T / 16;
  const int q0 = (tile % tilesPerHead) * 16;
  const int h  = (tile / tilesPerHead) % nh;
  const int b  = tile / (tilesPerHead * nh);

  const long ld = 3L * Cdim;
  const bf16* qb = qkv + (long)b * T * ld + (long)h * hd;
  const bf16* kb = qb + Cdim;
  const bf16* vb = qb + 2 * Cdim;

  const int mrow  = lane & 15;
  const int khalf = lane >> 4;

  // Q fragments for the two hd halves: two 16B runs per lane each
  FragU qF[2];
#pragma unroll
  for (int half = 0; half < 2; ++half) {
    const bf16* qr = qb + (long)(q0 + mrow) * ld + half * 32 + khalf * 16;
    qF[half].q[0] = *(const uint4*)(qr);
    qF[half].q[1] = *(const uint4*)(qr + 8);
  }

  v8f accO[4] = {};
  float mRun[8], lRun[8];
#pragma unroll
  for (int r = 0; r < 8; ++r) { mRun[r] = -1e30f; lRun[r] = 0.f; }

  const int nChunks = q0 / 32 + 1;       // covers kv rows [0, q0+15]
  for (int cIdx = 0; cIdx < nChunks; ++cIdx) {
    const int j0 = cIdx * 32;

    // S (16x32) = Q (16x64) * K^T : K^T fragment = 16 contiguous elems of a
    // K row (B-frag lane (n,kh) wants K[n][half*32 + kh*16 .. +15])
    v8f s[2] = {};
#pragma unroll
    for (int half = 0; half < 2; ++half) {
#pragma unroll
      for (int nt = 0; nt < 2; ++nt) {
        FragU bT;
        const bf16* kr =
            kb + (long)(j0 + nt * 16 + mrow) * ld + half * 32 + khalf * 16;
        bT.q[0] = *(const uint4*)(kr);
        bT.q[1] = *(const uint4*)(kr + 8);
        s[nt] = wmma_bf16(qF[half].v, bT.v, s[nt]);
      }
    }

    // scale + causal mask directly in accumulator layout
    const float scale = 0.125f;          // 1/sqrt(64)
#pragma unroll
    for (int r = 0; r < 8; ++r) {
      const int mg  = q0 + r + 8 * khalf;
      const int ng0 = j0 + mrow;
      float v0 = s[0][r] * scale;
      float v1 = s[1][r] * scale;
      if (ng0 > mg)      v0 = -1e30f;
      if (ng0 + 16 > mg) v1 = -1e30f;
      s[0][r] = v0; s[1][r] = v1;
    }

    // online softmax (row stats via half-wave shuffles), P -> LDS as bf16
#pragma unroll
    for (int r = 0; r < 8; ++r) {
      float mx = fmaxf(s[0][r], s[1][r]);
#pragma unroll
      for (int off = 1; off < 16; off <<= 1)
        mx = fmaxf(mx, __shfl_xor(mx, off, 16));
      const float mNew = fmaxf(mRun[r], mx);
      const float sc   = __expf(mRun[r] - mNew);
      const float p0   = __expf(s[0][r] - mNew);
      const float p1   = __expf(s[1][r] - mNew);
      float rs = p0 + p1;
#pragma unroll
      for (int off = 1; off < 16; off <<= 1)
        rs += __shfl_xor(rs, off, 16);
      lRun[r] = lRun[r] * sc + rs;
      mRun[r] = mNew;
#pragma unroll
      for (int t = 0; t < 4; ++t) accO[t][r] *= sc;
      const int m = r + 8 * khalf;
      Pb[wv][m][mrow]      = (bf16)p0;
      Pb[wv][m][mrow + 16] = (bf16)p1;
    }
    asm volatile("s_wait_dscnt 0" ::: "memory");  // LDS write->read, same wave

    // P (16x32) as an A fragment: two 16B LDS runs per lane
    FragU pF;
    {
      const bf16* pr = &Pb[wv][mrow][khalf * 16];
      pF.q[0] = *(const uint4*)(pr);
      pF.q[1] = *(const uint4*)(pr + 8);
    }

    // accO (16x64) += P (16x32) * V (32x64); V fragment is a column gather
#pragma unroll
    for (int t = 0; t < 4; ++t) {
      v16bf vF;
#pragma unroll
      for (int v = 0; v < 8; ++v) {
#pragma unroll
        for (int sL = 0; sL < 2; ++sL) {
          const int kk = khalf * 16 + 2 * v + sL;
          vF[2 * v + sL] = vb[(long)(j0 + kk) * ld + t * 16 + mrow];
        }
      }
      accO[t] = wmma_bf16(pF.v, vF, accO[t]);
    }
  }

  // normalize and store bf16 y
#pragma unroll
  for (int t = 0; t < 4; ++t) {
#pragma unroll
    for (int r = 0; r < 8; ++r) {
      const float v = accO[t][r] / lRun[r];
      const int m = q0 + r + 8 * khalf;
      const int n = h * hd + t * 16 + mrow;
      y[((long)b * T + m) * Cdim + n] = (bf16)v;
    }
  }
}

// ===========================================================================
// LayerNorm (population variance, eps 1e-5) -> bf16
// ===========================================================================
__global__ __launch_bounds__(256) void layernorm_kernel(
    const float* __restrict__ x, const float* __restrict__ g,
    bf16* __restrict__ out, int Cdim)
{
  const long n = blockIdx.x;
  __shared__ float redS[8], redQ[8];
  float s = 0.f, q = 0.f;
  for (int i = threadIdx.x; i < Cdim; i += 256) {
    const float v = x[n * Cdim + i];
    s += v; q += v * v;
  }
#pragma unroll
  for (int off = 16; off > 0; off >>= 1) {
    s += __shfl_xor(s, off, 32);
    q += __shfl_xor(q, off, 32);
  }
  if ((threadIdx.x & 31) == 0) { redS[threadIdx.x >> 5] = s; redQ[threadIdx.x >> 5] = q; }
  __syncthreads();
  s = 0.f; q = 0.f;
#pragma unroll
  for (int w = 0; w < 8; ++w) { s += redS[w]; q += redQ[w]; }
  const float mean = s / Cdim;
  const float var  = q / Cdim - mean * mean;
  const float inv  = rsqrtf(var + 1e-5f);
  for (int i = threadIdx.x; i < Cdim; i += 256)
    out[n * Cdim + i] = (bf16)((x[n * Cdim + i] - mean) * inv * g[i]);
}

// ===========================================================================
// MoE gating / routing / dispatch / combine
// ===========================================================================
__global__ void gate_logits_kernel(const bf16* __restrict__ tok,
                                   const float* __restrict__ wg,
                                   float* __restrict__ logits,
                                   int N, int Cdim, int E)
{
  const int idx = blockIdx.x * blockDim.x + threadIdx.x;
  if (idx >= N * E) return;
  const int n = idx / E, e = idx % E;
  float acc = 0.f;
  for (int c = 0; c < Cdim; ++c)
    acc += (float)tok[(long)n * Cdim + c] * wg[(long)c * E + e];
  logits[idx] = acc;
}

// Serial routing reproducing the reference's rank order: the cumsum runs over
// the [k, N] flattening, so all k=0 picks rank before any k=1 pick.
__global__ void route_kernel(const float* __restrict__ logits,
                             int* __restrict__ topi, int* __restrict__ slot,
                             float* __restrict__ wgt, int N, int E, int cap)
{
  if (threadIdx.x != 0 || blockIdx.x != 0) return;
  int cnt[8];
#pragma unroll
  for (int e = 0; e < 8; ++e) cnt[e] = 0;
  for (int n = 0; n < N; ++n) {              // pass 1: top-2 + k=0 ranks
    const float* L = logits + (long)n * E;
    int i1 = 0; float v1 = L[0];
    for (int e = 1; e < E; ++e) if (L[e] > v1) { v1 = L[e]; i1 = e; }
    int i2 = 0; float v2 = -3.4e38f;
    for (int e = 0; e < E; ++e) if (e != i1 && L[e] > v2) { v2 = L[e]; i2 = e; }
    const float mx = fmaxf(v1, v2);
    const float e1 = __expf(v1 - mx), e2 = __expf(v2 - mx);
    const float inv = 1.f / (e1 + e2);
    topi[2 * n] = i1; topi[2 * n + 1] = i2;
    wgt[2 * n] = e1 * inv; wgt[2 * n + 1] = e2 * inv;
    const int r = cnt[i1]++;
    slot[2 * n] = (r < cap) ? r : -1;
  }
  for (int n = 0; n < N; ++n) {              // pass 2: k=1 ranks
    const int i2 = topi[2 * n + 1];
    const int r = cnt[i2]++;
    slot[2 * n + 1] = (r < cap) ? r : -1;
  }
}

__global__ void zero_u4_kernel(uint4* __restrict__ p, long n16) {
  const uint4 z = {0u, 0u, 0u, 0u};
  for (long i = (long)blockIdx.x * blockDim.x + threadIdx.x; i < n16;
       i += (long)gridDim.x * blockDim.x)
    p[i] = z;
}

__global__ __launch_bounds__(256) void dispatch_kernel(
    const bf16* __restrict__ tok, const int* __restrict__ topi,
    const int* __restrict__ slot, bf16* __restrict__ disp,
    int Cdim, int cap)
{
  const int n = blockIdx.x >> 1, kk = blockIdx.x & 1;
  const int s = slot[2 * n + kk];
  if (s < 0) return;
  const int e = topi[2 * n + kk];
  const uint2* src = (const uint2*)(tok + (long)n * Cdim);
  uint2* dst = (uint2*)(disp + ((long)e * cap + s) * Cdim);
  for (int i = threadIdx.x; i < Cdim / 4; i += 256) dst[i] = src[i];
}

__global__ __launch_bounds__(256) void combine_kernel(
    const float* __restrict__ x2, const float* __restrict__ eout,
    const int* __restrict__ topi, const int* __restrict__ slot,
    const float* __restrict__ wgt, float* __restrict__ out,
    int Cdim, int cap)
{
  const long n = blockIdx.x;
  const int e0 = topi[2 * n], e1 = topi[2 * n + 1];
  const int s0 = slot[2 * n], s1 = slot[2 * n + 1];
  const float w0 = wgt[2 * n], w1 = wgt[2 * n + 1];
  const float* p0 = (s0 >= 0) ? eout + ((long)e0 * cap + s0) * Cdim : nullptr;
  const float* p1 = (s1 >= 0) ? eout + ((long)e1 * cap + s1) * Cdim : nullptr;
  for (int i = threadIdx.x; i < Cdim; i += 256) {
    float v = x2[n * Cdim + i];
    if (p0) v += w0 * p0[i];
    if (p1) v += w1 * p1[i];
    out[n * Cdim + i] = v;
  }
}

__global__ void cast_f32_bf16_kernel(const float* __restrict__ in,
                                     bf16* __restrict__ out, long n) {
  for (long i = (long)blockIdx.x * blockDim.x + threadIdx.x; i < n;
       i += (long)gridDim.x * blockDim.x)
    out[i] = (bf16)in[i];
}

// ===========================================================================
extern "C" void kernel_launch(void* const* d_in, const int* in_sizes, int n_in,
                              void* d_out, int out_size, void* d_ws, size_t ws_size,
                              hipStream_t stream)
{
  const int Bb = 4, T = 1024, Cdim = 1024, E = 8, H = 4096, NH = 16;
  const int NT  = Bb * T;          // 4096 tokens
  const int CAP = 1280;            // floor(2*1.25*NT/E), even, >= MIN_CAPACITY

  const float* x      = (const float*)d_in[0];
  const float* g1     = (const float*)d_in[1];
  const float* g2     = (const float*)d_in[2];
  const float* w_attn = (const float*)d_in[3];
  const float* w_proj = (const float*)d_in[4];
  const float* w_g    = (const float*)d_in[5];
  const float* c_fc   = (const float*)d_in[6];
  const float* c_proj = (const float*)d_in[7];
  // d_in[8] = top_k (== 2, baked into the routing path)

  char* ws = (char*)d_ws;
  auto alloc = [&](size_t bytes) -> char* {
    char* p = ws;
    ws += (bytes + 255) & ~(size_t)255;
    return p;
  };
  bf16*  ln1     = (bf16*) alloc((size_t)NT * Cdim * 2);
  bf16*  wattn_b = (bf16*) alloc((size_t)Cdim * 3 * Cdim * 2);
  bf16*  qkv     = (bf16*) alloc((size_t)NT * 3 * Cdim * 2);
  bf16*  ybuf    = (bf16*) alloc((size_t)NT * Cdim * 2);
  bf16*  wproj_b = (bf16*) alloc((size_t)Cdim * Cdim * 2);
  float* x2      = (float*)alloc((size_t)NT * Cdim * 4);
  bf16*  ln2     = (bf16*) alloc((size_t)NT * Cdim * 2);
  float* logits  = (float*)alloc((size_t)NT * E * 4);
  int*   topi    = (int*)  alloc((size_t)NT * 2 * 4);
  int*   slot    = (int*)  alloc((size_t)NT * 2 * 4);
  float* wgt     = (float*)alloc((size_t)NT * 2 * 4);
  bf16*  disp    = (bf16*) alloc((size_t)E * CAP * Cdim * 2);
  bf16*  cfc_b   = (bf16*) alloc((size_t)E * Cdim * H * 2);
  bf16*  cproj_b = (bf16*) alloc((size_t)E * H * Cdim * 2);
  bf16*  hmid    = (bf16*) alloc((size_t)E * CAP * H * 2);
  float* eout    = (float*)alloc((size_t)E * CAP * Cdim * 4);

  // weight down-conversions (deterministic, done every call)
  cast_f32_bf16_kernel<<<1024, 256, 0, stream>>>(w_attn, wattn_b, (long)Cdim * 3 * Cdim);
  cast_f32_bf16_kernel<<<1024, 256, 0, stream>>>(w_proj, wproj_b, (long)Cdim * Cdim);
  cast_f32_bf16_kernel<<<2048, 256, 0, stream>>>(c_fc,   cfc_b,   (long)E * Cdim * H);
  cast_f32_bf16_kernel<<<2048, 256, 0, stream>>>(c_proj, cproj_b, (long)E * H * Cdim);

  // ---- attention branch ----
  layernorm_kernel<<<NT, 256, 0, stream>>>(x, g1, ln1, Cdim);

  dim3 g_qkv(3 * Cdim / 128, NT / 64, 1);
  gemm_bf16_wmma_kernel<<<g_qkv, 256, GEMM_LDS_BYTES, stream>>>(
      ln1, wattn_b, qkv, nullptr, NT, 3 * Cdim, Cdim, 0, 0, 0, 0, /*bf16 out*/2);

  attn_kernel<<<(Bb * NH * (T / 16)) / 8, 256, 0, stream>>>(qkv, ybuf, T, Cdim, NH);

  dim3 g_proj(Cdim / 128, NT / 64, 1);
  gemm_bf16_wmma_kernel<<<g_proj, 256, GEMM_LDS_BYTES, stream>>>(
      ybuf, wproj_b, x2, x, NT, Cdim, Cdim, 0, 0, 0, 0, /*+residual, f32 out*/4);

  // ---- MoE branch ----
  layernorm_kernel<<<NT, 256, 0, stream>>>(x2, g2, ln2, Cdim);

  gate_logits_kernel<<<(NT * E + 255) / 256, 256, 0, stream>>>(ln2, w_g, logits, NT, Cdim, E);
  route_kernel<<<1, 32, 0, stream>>>(logits, topi, slot, wgt, NT, E, CAP);

  zero_u4_kernel<<<2048, 256, 0, stream>>>((uint4*)disp, (long)E * CAP * Cdim * 2 / 16);
  dispatch_kernel<<<NT * 2, 256, 0, stream>>>(ln2, topi, slot, disp, Cdim, CAP);

  dim3 g_fc(H / 128, CAP / 64, E);
  gemm_bf16_wmma_kernel<<<g_fc, 256, GEMM_LDS_BYTES, stream>>>(
      disp, cfc_b, hmid, nullptr, CAP, H, Cdim,
      (long)CAP * Cdim, (long)Cdim * H, (long)CAP * H, 0, /*gelu + bf16 out*/3);

  dim3 g_pr(Cdim / 128, CAP / 64, E);
  gemm_bf16_wmma_kernel<<<g_pr, 256, GEMM_LDS_BYTES, stream>>>(
      hmid, cproj_b, eout, nullptr, CAP, Cdim, H,
      (long)CAP * H, (long)H * Cdim, (long)CAP * Cdim, 0, /*f32 out*/0);

  combine_kernel<<<NT, 256, 0, stream>>>(x2, eout, topi, slot, wgt,
                                         (float*)d_out, Cdim, CAP);
}
